// NodeEdgeAttBlock_27084063768593
// MI455X (gfx1250) — matrix operand order repaired and checked by reading, not compiled, not run on hardware
//
#include <hip/hip_runtime.h>
#include <hip/hip_bf16.h>

#define BSZ 2
#define NN  256
#define NHN 256
#define NHE 128
#define NHY 128

typedef __attribute__((ext_vector_type(16))) _Float16 v16h;
typedef __attribute__((ext_vector_type(8)))  float    v8f;

// ---------------- workspace layout (float element offsets) ----------------
constexpr size_t OFF_Q   = 0;                 // 131072
constexpr size_t OFF_K   = OFF_Q  + 131072;   // 131072
constexpr size_t OFF_V   = OFF_K  + 131072;   // 131072
constexpr size_t OFF_WV  = OFF_V  + 131072;   // 131072
constexpr size_t OFF_YE1 = OFF_WV + 131072;   // 512
constexpr size_t OFF_YE2 = OFF_YE1 + 512;     // 512
constexpr size_t OFF_YX1 = OFF_YE2 + 512;     // 512
constexpr size_t OFF_YX2 = OFF_YX1 + 512;     // 512
constexpr size_t OFF_ZX  = OFF_YX2 + 512;     // 2048 (2 x 1024)
constexpr size_t OFF_ZA  = OFF_ZX  + 2048;    // 1024 (2 x 512)
constexpr size_t OFF_F16 = OFF_ZA  + 1024;    // f16 region starts here (32B aligned)

__device__ __forceinline__ v8f wmma_f16(v16h a, v16h b, v8f c) {
  // D = A(16x32 f16) x B(32x16 f16) + C(16x16 f32)
  return __builtin_amdgcn_wmma_f32_16x16x32_f16(false, a, false, b, (short)0, c,
                                                false, false);
}

// ---- pre-swizzle f32 weights into per-lane v16h WMMA B fragments ----------
// B layout (32x16 f16): lane = n + 16*h; half e of lane holds
// K = (lane>>4)*8 + (e&7) + (e>>3)*16 of column n (ISA table order).
__global__ void k_prep_weights(const float* __restrict__ w1,
                               const float* __restrict__ w2,
                               const float* __restrict__ wo,
                               _Float16* __restrict__ w1sw,
                               _Float16* __restrict__ w2sw,
                               _Float16* __restrict__ wosw) {
  int id = blockIdx.x * 256 + threadIdx.x;            // 0..98303
  if (id < 65536) {                                   // a2x weights: K=128, N=256
    const float* src = (id < 32768) ? w1 : w2;
    _Float16*    dst = (id < 32768) ? w1sw : w2sw;
    int m     = id & 32767;
    int chunk = m >> 13;            // 16 ntiles * 512 = 8192 per K-chunk
    int rem   = m & 8191;
    int ntile = rem >> 9;
    int lane  = (rem >> 4) & 31;
    int e     = rem & 15;
    int k     = chunk * 32 + (lane >> 4) * 8 + (e & 7) + (e >> 3) * 16;
    int n     = ntile * 16 + (lane & 15);
    dst[m] = (_Float16)src[k * NHN + n];
  } else {                                            // a_out_w: K=256, N=128
    int m     = id - 65536;
    int chunk = m >> 12;            // 8 ntiles * 512 = 4096 per K-chunk
    int rem   = m & 4095;
    int ntile = rem >> 9;
    int lane  = (rem >> 4) & 31;
    int e     = rem & 15;
    int k     = chunk * 32 + (lane >> 4) * 8 + (e & 7) + (e >> 3) * 16;
    int n     = ntile * 16 + (lane & 15);
    wosw[m] = (_Float16)wo[k * NHE + n];
  }
}

// ---- Q/K/V projections (small: 512x256x256 each) --------------------------
__global__ void k_prep_qkv(const float* __restrict__ x,
                           const float* __restrict__ qw, const float* __restrict__ qb,
                           const float* __restrict__ kw, const float* __restrict__ kb,
                           const float* __restrict__ vw, const float* __restrict__ vb,
                           float* __restrict__ Qf, float* __restrict__ Kf,
                           float* __restrict__ Vf) {
  __shared__ float xs[NHN];
  int row = blockIdx.x, c = threadIdx.x;
  xs[c] = x[row * NHN + c];
  __syncthreads();
  float q = qb[c], k = kb[c], v = vb[c];
  for (int t = 0; t < NHN; t++) {
    float xv = xs[t];
    q += xv * qw[t * NHN + c];
    k += xv * kw[t * NHN + c];
    v += xv * vw[t * NHN + c];
  }
  Qf[row * NHN + c] = q;
  Kf[row * NHN + c] = k;
  Vf[row * NHN + c] = v;
}

// ---- y-conditioning vectors ----------------------------------------------
__global__ void k_prep_yvecs(const float* __restrict__ y,
                             const float* __restrict__ ea_w, const float* __restrict__ ea_b,
                             const float* __restrict__ em_w, const float* __restrict__ em_b,
                             const float* __restrict__ xa_w, const float* __restrict__ xa_b,
                             const float* __restrict__ xm_w, const float* __restrict__ xm_b,
                             float* __restrict__ ye1, float* __restrict__ ye2,
                             float* __restrict__ yx1, float* __restrict__ yx2) {
  __shared__ float ys[NHY];
  int b = blockIdx.x, c = threadIdx.x;
  if (c < NHY) ys[c] = y[b * NHY + c];
  __syncthreads();
  float s1 = ea_b[c], s2 = em_b[c], s3 = xa_b[c], s4 = xm_b[c];
  for (int t = 0; t < NHY; t++) {
    float yv = ys[t];
    s1 += yv * ea_w[t * NHN + c];
    s2 += yv * em_w[t * NHN + c];
    s3 += yv * xa_w[t * NHN + c];
    s4 += yv * xm_w[t * NHN + c];
  }
  ye1[b * NHN + c] = s1;
  ye2[b * NHN + c] = s2;
  yx1[b * NHN + c] = s3;
  yx2[b * NHN + c] = s4;
}

// ---- main fused kernel: edge GEMMs + Y + online softmax + ahat GEMM -------
// one workgroup (8 waves) per (b,i); loop over 16-row j tiles.
__global__ void __launch_bounds__(256) k_fused(
    const float* __restrict__ a,
    const float* __restrict__ Qf, const float* __restrict__ Kf,
    const float* __restrict__ Vf,
    const _Float16* __restrict__ w1sw, const _Float16* __restrict__ w2sw,
    const _Float16* __restrict__ wosw,
    const float* __restrict__ a1bias, const float* __restrict__ a2bias,
    const float* __restrict__ aobias,
    const float* __restrict__ ye1, const float* __restrict__ ye2,
    float* __restrict__ wV, float* __restrict__ ahat) {
  __shared__ __align__(32) _Float16 Asw[4 * 512];   // a tile, A-fragment swizzled
  __shared__ __align__(32) _Float16 Zsw[8 * 512];   // Z tile, A-fragment swizzled

  const int tid  = threadIdx.x;
  const int wave = tid >> 5, lane = tid & 31;
  const int low  = lane & 15, hi = lane >> 4;
  const int b    = blockIdx.x / NN, i = blockIdx.x % NN;
  const float s  = 0.17677669529663688f;            // 1/sqrt(F), F=32

  const int c0 = wave * 32 + low;                   // this wave's channel cols
  const int c1 = c0 + 16;
  const float qv0 = Qf[(b * NN + i) * NHN + c0];
  const float qv1 = Qf[(b * NN + i) * NHN + c1];
  const float e10 = ye1[b * NHN + c0], e11 = ye1[b * NHN + c1];
  const float e20 = ye2[b * NHN + c0] + 1.0f, e21 = ye2[b * NHN + c1] + 1.0f;
  const float b1a0 = a1bias[c0], b1a1 = a1bias[c1];
  const float b2a0 = a2bias[c0], b2a1 = a2bias[c1];
  const float boE  = aobias[wave * 16 + low];

  float m0 = -3.0e38f, m1 = -3.0e38f;               // online softmax state
  float l0 = 0.f, l1 = 0.f, ac0 = 0.f, ac1 = 0.f;
  const size_t abase = (size_t)(b * NN + i) * NN * NHE;

  for (int j0 = 0; j0 < NN; j0 += 16) {
    // (a) stage a[b,i,j0:j0+16,0:128] as f16 A-fragments into LDS
    #pragma unroll
    for (int q = 0; q < 8; q++) {
      int pos = tid + q * 256;                      // 2048 fragment slots
      int chunk = pos >> 9;
      int l2 = (pos >> 4) & 31;
      int e  = pos & 15;
      int g  = l2 >> 4, mrow = l2 & 15;
      int kl = e + 8 * g + ((e >= 8) ? 8 : 0);      // ISA 16-bit A layout
      Asw[pos] = (_Float16)a[abase + (size_t)(j0 + mrow) * NHE + chunk * 32 + kl];
    }
    __syncthreads();

    // (b) GEMM1: a1/a2 tiles (16 x 32 per wave), K=128 over 4 WMMA chunks
    v8f c1a, c1b, c2a, c2b;
    #pragma unroll
    for (int r = 0; r < 8; r++) { c1a[r] = b1a0; c1b[r] = b1a1;
                                  c2a[r] = b2a0; c2b[r] = b2a1; }
    #pragma unroll
    for (int kc = 0; kc < 4; kc++) {
      v16h af = *(const v16h*)(&Asw[kc * 512 + lane * 16]);
      v16h bf;
      bf = *(const v16h*)(&w1sw[((kc * 16 + wave * 2    ) * 32 + lane) * 16]);
      c1a = wmma_f16(af, bf, c1a);
      bf = *(const v16h*)(&w1sw[((kc * 16 + wave * 2 + 1) * 32 + lane) * 16]);
      c1b = wmma_f16(af, bf, c1b);
      bf = *(const v16h*)(&w2sw[((kc * 16 + wave * 2    ) * 32 + lane) * 16]);
      c2a = wmma_f16(af, bf, c2a);
      bf = *(const v16h*)(&w2sw[((kc * 16 + wave * 2 + 1) * 32 + lane) * 16]);
      c2b = wmma_f16(af, bf, c2b);
    }

    // (c) Y = q*k*s*(1+a1)+a2 ; stage Z = ye1+(ye2+1)*Y ; online softmax
    float yv0[8], yv1[8];
    float tmax0 = -3.0e38f, tmax1 = -3.0e38f;
    #pragma unroll
    for (int r = 0; r < 8; r++) {
      int j = j0 + r + 8 * hi;                      // D row = r + 8*(lane>=16)
      float kv0 = Kf[(b * NN + j) * NHN + c0];
      float kv1 = Kf[(b * NN + j) * NHN + c1];
      yv0[r] = qv0 * kv0 * s * (1.f + c1a[r]) + c2a[r];
      yv1[r] = qv1 * kv1 * s * (1.f + c1b[r]) + c2b[r];
      tmax0 = fmaxf(tmax0, yv0[r]);
      tmax1 = fmaxf(tmax1, yv1[r]);
      int laneA = (r + 8 * hi) + 16 * ((low >> 3) & 1);
      Zsw[wave * 512 + laneA * 16 + (low & 7)    ] = (_Float16)(e10 + e20 * yv0[r]);
      Zsw[wave * 512 + laneA * 16 + (low & 7) + 8] = (_Float16)(e11 + e21 * yv1[r]);
    }
    float M0 = fmaxf(m0, tmax0), M1 = fmaxf(m1, tmax1);
    float r0 = __expf(m0 - M0), r1 = __expf(m1 - M1);
    l0 *= r0; ac0 *= r0; l1 *= r1; ac1 *= r1;
    #pragma unroll
    for (int r = 0; r < 8; r++) {
      int j = j0 + r + 8 * hi;
      float p0 = __expf(yv0[r] - M0), p1 = __expf(yv1[r] - M1);
      l0 += p0; l1 += p1;
      ac0 += p0 * Vf[(b * NN + j) * NHN + c0];
      ac1 += p1 * Vf[(b * NN + j) * NHN + c1];
    }
    m0 = M0; m1 = M1;
    __syncthreads();

    // (d) GEMM2: ahat tile = Z(16x256) @ a_out_w(256x128); wave -> 16 cols
    v8f cE;
    #pragma unroll
    for (int r = 0; r < 8; r++) cE[r] = boE;
    #pragma unroll
    for (int kc = 0; kc < 8; kc++) {
      v16h af = *(const v16h*)(&Zsw[kc * 512 + lane * 16]);
      v16h bf = *(const v16h*)(&wosw[((kc * 8 + wave) * 32 + lane) * 16]);
      cE = wmma_f16(af, bf, cE);
    }
    #pragma unroll
    for (int r = 0; r < 8; r++) {
      int j = j0 + r + 8 * hi;
      ahat[((size_t)(b * NN + i) * NN + j) * NHE + wave * 16 + low] = cE[r];
    }
    __syncthreads();
  }

  // merge softmax state across the lane^16 pair (rows 0-7 vs 8-15) -> wV
  float pm0 = __shfl_xor(m0, 16, 32), pl0 = __shfl_xor(l0, 16, 32);
  float pa0 = __shfl_xor(ac0, 16, 32);
  float pm1 = __shfl_xor(m1, 16, 32), pl1 = __shfl_xor(l1, 16, 32);
  float pa1 = __shfl_xor(ac1, 16, 32);
  float M0 = fmaxf(m0, pm0), M1 = fmaxf(m1, pm1);
  float L0 = l0 * __expf(m0 - M0) + pl0 * __expf(pm0 - M0);
  float L1 = l1 * __expf(m1 - M1) + pl1 * __expf(pm1 - M1);
  float A0 = ac0 * __expf(m0 - M0) + pa0 * __expf(pm0 - M0);
  float A1 = ac1 * __expf(m1 - M1) + pa1 * __expf(pm1 - M1);
  if (hi == 0) {
    wV[(b * NN + i) * NHN + c0] = A0 / L0;
    wV[(b * NN + i) * NHN + c1] = A1 / L1;
  }
}

// ---- xhat = lin(yx1 + (yx2+1)*wV, x_out_w) --------------------------------
__global__ void k_xhat(const float* __restrict__ wV,
                       const float* __restrict__ yx1, const float* __restrict__ yx2,
                       const float* __restrict__ xow, const float* __restrict__ xob,
                       float* __restrict__ out) {
  __shared__ float u[NHN];
  int row = blockIdx.x, b = row / NN, c = threadIdx.x;
  u[c] = yx1[b * NHN + c] + (yx2[b * NHN + c] + 1.f) * wV[row * NHN + c];
  __syncthreads();
  float o = xob[c];
  for (int t = 0; t < NHN; t++) o += u[t] * xow[t * NHN + c];
  out[row * NHN + c] = o;
}

// ---- zx stats (mean/min/max/std ddof=1 over axis 1) -----------------------
__global__ void k_stats_x(const float* __restrict__ x, float* __restrict__ zx) {
  int b = blockIdx.x, c = threadIdx.x;
  float sum = 0.f, sq = 0.f, mn = 3.0e38f, mx = -3.0e38f;
  for (int i2 = 0; i2 < NN; i2++) {
    float v = x[(b * NN + i2) * NHN + c];
    sum += v; sq += v * v; mn = fminf(mn, v); mx = fmaxf(mx, v);
  }
  float mean = sum / (float)NN;
  float var  = (sq - sum * sum / (float)NN) / (float)(NN - 1);
  zx[b * 4 * NHN + c]            = mean;
  zx[b * 4 * NHN + NHN + c]      = mn;
  zx[b * 4 * NHN + 2 * NHN + c]  = mx;
  zx[b * 4 * NHN + 3 * NHN + c]  = sqrtf(fmaxf(var, 0.f));
}

// ---- za stats over axes (1,2) of a ----------------------------------------
__global__ void k_stats_a(const float* __restrict__ a, float* __restrict__ za) {
  __shared__ float ssum[256], ssq[256], smn[256], smx[256];
  int b = blockIdx.x >> 7, c = blockIdx.x & 127;
  int tid = threadIdx.x;
  float sum = 0.f, sq = 0.f, mn = 3.0e38f, mx = -3.0e38f;
  size_t base = (size_t)b * NN * NN * NHE + c;
  for (int t = 0; t < 256; t++) {
    float v = a[base + (size_t)(t * 256 + tid) * NHE];
    sum += v; sq += v * v; mn = fminf(mn, v); mx = fmaxf(mx, v);
  }
  ssum[tid] = sum; ssq[tid] = sq; smn[tid] = mn; smx[tid] = mx;
  __syncthreads();
  for (int st = 128; st > 0; st >>= 1) {
    if (tid < st) {
      ssum[tid] += ssum[tid + st];
      ssq[tid]  += ssq[tid + st];
      smn[tid]   = fminf(smn[tid], smn[tid + st]);
      smx[tid]   = fmaxf(smx[tid], smx[tid + st]);
    }
    __syncthreads();
  }
  if (tid == 0) {
    const float cnt = 65536.f;
    float mean = ssum[0] / cnt;
    float var  = (ssq[0] - ssum[0] * ssum[0] / cnt) / (cnt - 1.f);
    za[b * 4 * NHE + c]           = mean;
    za[b * 4 * NHE + NHE + c]     = smn[0];
    za[b * 4 * NHE + 2 * NHE + c] = smx[0];
    za[b * 4 * NHE + 3 * NHE + c] = sqrtf(fmaxf(var, 0.f));
  }
}

// ---- final yhat MLP -------------------------------------------------------
__global__ void k_final_y(const float* __restrict__ y,
                          const float* __restrict__ zx, const float* __restrict__ za,
                          const float* __restrict__ yyw, const float* __restrict__ yyb,
                          const float* __restrict__ xyw, const float* __restrict__ xyb,
                          const float* __restrict__ ayw, const float* __restrict__ ayb,
                          const float* __restrict__ o1w, const float* __restrict__ o1b,
                          const float* __restrict__ o2w, const float* __restrict__ o2b,
                          float* __restrict__ out) {
  __shared__ float buf[NHY];
  int b = blockIdx.x, c = threadIdx.x;
  float h = yyb[c] + xyb[c] + ayb[c];
  for (int t = 0; t < NHY; t++)     h += y[b * NHY + t]      * yyw[t * NHY + c];
  for (int t = 0; t < 4 * NHE; t++) h += za[b * 4 * NHE + t] * ayw[t * NHY + c];
  for (int t = 0; t < 4 * NHN; t++) h += zx[b * 4 * NHN + t] * xyw[t * NHY + c];
  buf[c] = h;
  __syncthreads();
  float h2 = o1b[c];
  for (int t = 0; t < NHY; t++) h2 += buf[t] * o1w[t * NHY + c];
  h2 = fmaxf(h2, 0.f);
  __syncthreads();
  buf[c] = h2;
  __syncthreads();
  float o = o2b[c];
  for (int t = 0; t < NHY; t++) o += buf[t] * o2w[t * NHY + c];
  out[b * NHY + c] = o;
}

extern "C" void kernel_launch(void* const* d_in, const int* in_sizes, int n_in,
                              void* d_out, int out_size, void* d_ws, size_t ws_size,
                              hipStream_t stream) {
  const float* x         = (const float*)d_in[0];
  const float* a         = (const float*)d_in[1];
  const float* y         = (const float*)d_in[2];
  const float* qw        = (const float*)d_in[3];
  const float* qb        = (const float*)d_in[4];
  const float* kw        = (const float*)d_in[5];
  const float* kb        = (const float*)d_in[6];
  const float* vw        = (const float*)d_in[7];
  const float* vb        = (const float*)d_in[8];
  const float* a2x_add_w = (const float*)d_in[9];
  const float* a2x_add_b = (const float*)d_in[10];
  const float* a2x_mul_w = (const float*)d_in[11];
  const float* a2x_mul_b = (const float*)d_in[12];
  const float* y2e_mul_w = (const float*)d_in[13];
  const float* y2e_mul_b = (const float*)d_in[14];
  const float* y2e_add_w = (const float*)d_in[15];
  const float* y2e_add_b = (const float*)d_in[16];
  const float* y2x_mul_w = (const float*)d_in[17];
  const float* y2x_mul_b = (const float*)d_in[18];
  const float* y2x_add_w = (const float*)d_in[19];
  const float* y2x_add_b = (const float*)d_in[20];
  const float* y_y_w     = (const float*)d_in[21];
  const float* y_y_b     = (const float*)d_in[22];
  const float* x_y_w     = (const float*)d_in[23];
  const float* x_y_b     = (const float*)d_in[24];
  const float* a_y_w     = (const float*)d_in[25];
  const float* a_y_b     = (const float*)d_in[26];
  const float* x_out_w   = (const float*)d_in[27];
  const float* x_out_b   = (const float*)d_in[28];
  const float* a_out_w   = (const float*)d_in[29];
  const float* a_out_b   = (const float*)d_in[30];
  const float* y_out1_w  = (const float*)d_in[31];
  const float* y_out1_b  = (const float*)d_in[32];
  const float* y_out2_w  = (const float*)d_in[33];
  const float* y_out2_b  = (const float*)d_in[34];

  float* ws   = (float*)d_ws;
  float* Qf   = ws + OFF_Q;
  float* Kf   = ws + OFF_K;
  float* Vf   = ws + OFF_V;
  float* wVp  = ws + OFF_WV;
  float* ye1  = ws + OFF_YE1;
  float* ye2  = ws + OFF_YE2;
  float* yx1  = ws + OFF_YX1;
  float* yx2  = ws + OFF_YX2;
  float* zx   = ws + OFF_ZX;
  float* za   = ws + OFF_ZA;
  _Float16* w1sw = (_Float16*)(ws + OFF_F16);
  _Float16* w2sw = w1sw + 32768;
  _Float16* wosw = w2sw + 32768;

  float* out_xhat = (float*)d_out;                       // 2*256*256
  float* out_ahat = out_xhat + (size_t)BSZ * NN * NN;    // 2*256*256*128
  float* out_yhat = out_ahat + (size_t)BSZ * NN * NN * NHE;

  k_prep_weights<<<384, 256, 0, stream>>>(a2x_add_w, a2x_mul_w, a_out_w,
                                          w1sw, w2sw, wosw);
  k_prep_qkv<<<BSZ * NN, 256, 0, stream>>>(x, qw, qb, kw, kb, vw, vb, Qf, Kf, Vf);
  k_prep_yvecs<<<BSZ, 256, 0, stream>>>(y, y2e_add_w, y2e_add_b, y2e_mul_w,
                                        y2e_mul_b, y2x_add_w, y2x_add_b,
                                        y2x_mul_w, y2x_mul_b, ye1, ye2, yx1, yx2);
  k_fused<<<BSZ * NN, 256, 0, stream>>>(a, Qf, Kf, Vf, w1sw, w2sw, wosw,
                                        a2x_add_b, a2x_mul_b, a_out_b,
                                        ye1, ye2, wVp, out_ahat);
  k_xhat<<<BSZ * NN, 256, 0, stream>>>(wVp, yx1, yx2, x_out_w, x_out_b, out_xhat);
  k_stats_x<<<BSZ, 256, 0, stream>>>(x, zx);
  k_stats_a<<<BSZ * NHE, 256, 0, stream>>>(a, za);
  k_final_y<<<BSZ, NHY, 0, stream>>>(y, zx, za, y_y_w, y_y_b, x_y_w, x_y_b,
                                     a_y_w, a_y_b, y_out1_w, y_out1_b,
                                     y_out2_w, y_out2_b, out_yhat);
}
